// RNNModel_58153857188093
// MI455X (gfx1250) — compile-verified, matrix operand order checked by reading, not run on hardware
//
#include <hip/hip_runtime.h>
#include <hip/hip_bf16.h>

// ---------------- dims ----------------
#define T_    256
#define B_    128
#define NTOK_ 1024
#define NINP_ 600
#define NHID_ 1200
#define NB_   6
#define BS_   200
#define BSP_  224     // BS padded to multiple of 32
#define DK_   64
#define ATT_  512
#define G4_   800     // 4*BS
#define HC_   4
#define DKC_  32
#define DVC_  32

typedef __attribute__((ext_vector_type(16))) __bf16 v16bf;
typedef __attribute__((ext_vector_type(8)))  float  v8f;
typedef __attribute__((ext_vector_type(8)))  __bf16 bf16x8;
typedef __attribute__((ext_vector_type(4)))  unsigned u32x4;
typedef __attribute__((ext_vector_type(4)))  int      i32x4v;
typedef __attribute__((ext_vector_type(8)))  int      i32x8v;

#define LDSK 40   // 32 K-columns + 8 pad (80B row stride; TDM pad reproduces this)

// ---- Tensor Data Mover availability (probe-verified builtins) ----
#if defined(__has_builtin)
#if __has_builtin(__builtin_amdgcn_tensor_load_to_lds) && __has_builtin(__builtin_amdgcn_s_wait_tensorcnt)
#define HAVE_TDM 1
#endif
#endif
#ifndef HAVE_TDM
#define HAVE_TDM 0
#endif

// ---------------------------------------------------------------------------
// TDM: issue a 2D tile load (128 rows x 32 bf16 cols) global -> LDS with
// hardware row padding (16 DWORDs data + 4 DWORDs pad == LDSK*2 = 80B rows).
// D# layout per CDNA5 ISA. Rows/cols beyond the tensor
// dims zero-fill in LDS (hardware OOB behavior), so no guards needed.
// ---------------------------------------------------------------------------
#if HAVE_TDM
__device__ inline void tdm_load_tile_2d(const void* lds_dst, const void* gsrc,
                                        int rows_remaining, int k_remaining,
                                        long row_stride_elems)
{
    const unsigned long long ga = (unsigned long long)gsrc;
    const unsigned lds_addr = (unsigned)(unsigned long long)lds_dst; // low32 = LDS offset

    u32x4 g0;
    g0[0] = 1u;                                           // count=1, user mode
    g0[1] = lds_addr;                                     // lds_addr
    g0[2] = (unsigned)(ga & 0xffffffffu);                 // global_addr[31:0]
    g0[3] = (unsigned)((ga >> 32) & 0x01ffffffu) | (2u << 30); // addr[56:32] | type=2

    const unsigned dim0 = (unsigned)k_remaining;          // tensor_dim0 (elems)
    const unsigned dim1 = (unsigned)rows_remaining;       // tensor_dim1 (rows)
    const unsigned long long s0 = (unsigned long long)row_stride_elems;

    i32x8v g1;
    // data_size=2B (code 1), pad_enable, pad_interval=16 DW (code 3),
    // pad_amount=4 DW (code 3), no multicast / iterate / barrier.
    g1[0] = (int)((1u << 16) | (1u << 20) | (3u << 22) | (3u << 25));
    g1[1] = (int)((dim0 & 0xffffu) << 16);                // tensor_dim0[15:0]
    g1[2] = (int)((dim0 >> 16) | ((dim1 & 0xffffu) << 16)); // dim0[31:16]|dim1[15:0]
    g1[3] = (int)((dim1 >> 16) | (32u << 16));            // dim1[31:16] | tile_dim0=32
    g1[4] = (int)128u;                                    // tile_dim1=128, tile_dim2=0
    g1[5] = (int)(s0 & 0xffffffffu);                      // dim0_stride[31:0]
    g1[6] = (int)((s0 >> 32) & 0xffffu);                  // dim0_stride[47:32]
    g1[7] = 0;

    const i32x4v z4 = {0, 0, 0, 0};
#if __clang_major__ >= 23
    const i32x8v z8 = {0, 0, 0, 0, 0, 0, 0, 0};
    __builtin_amdgcn_tensor_load_to_lds(g0, g1, z4, z4, z8, 0);
#else
    __builtin_amdgcn_tensor_load_to_lds(g0, g1, z4, z4, 0);
#endif
}
#endif

// ---------------------------------------------------------------------------
// WMMA fragment load from an LDS tile stored row-major [row][LDSK] (bf16).
// 16-bit A/B layout (ISA 7.12.2): lanes 0-15 -> K 0..7 & 16..23, lanes
// 16-31 -> K 8..15 & 24..31, element row = lane&15 (B tile is N-major).
// ---------------------------------------------------------------------------
__device__ inline v16bf load_frag(const __bf16* tile, int rowbase, int lane) {
    const int r  = rowbase + (lane & 15);
    const int kb = (lane >> 4) << 3;
    const __bf16* p = tile + (long)r * LDSK;
    v16bf f;
#pragma unroll
    for (int i = 0; i < 8; ++i) {
        f[i]     = p[kb + i];
        f[i + 8] = p[kb + 16 + i];
    }
    return f;
}

// ---------------------------------------------------------------------------
// Core: C(128x128 tile) += A(MxK) * Bt(NxK)^T. A is bf16 (ABF16=true) or f32.
// Bt is always pre-transposed bf16 [N][Kpad]. K must be a multiple of 32.
// 8 waves: 4(M) x 2(N); each wave owns 32x64 = 2x4 16x16 f32 accumulators.
// Staging via TDM (one issuing wave, s_wait_tensorcnt) when available.
// ---------------------------------------------------------------------------
template<bool ABF16>
__device__ inline void gemm_tiles_accum(
    const void* __restrict__ A, int lda, int M, int arowbase,
    const __bf16* __restrict__ Bt, int ldbt, int N, int bcolbase,
    int K,
    __bf16 (*As)[LDSK], __bf16 (*Bs)[LDSK],
    v8f acc[2][4], int tid)
{
    const int lane = tid & 31;
    const int w    = tid >> 5;
    const int wm   = (w & 3) * 32;
    const int wn   = (w >> 2) * 64;

    for (int k0 = 0; k0 < K; k0 += 32) {
#if HAVE_TDM
        const bool issuer = (tid >> 5) == 0;   // wave 0 drives the TDM
        if (issuer) {
            tdm_load_tile_2d(&Bs[0][0], Bt + (size_t)bcolbase * ldbt + k0,
                             N - bcolbase, K - k0, ldbt);
            if constexpr (ABF16)
                tdm_load_tile_2d(&As[0][0],
                                 (const __bf16*)A + (size_t)arowbase * lda + k0,
                                 M - arowbase, K - k0, lda);
        }
        if constexpr (!ABF16) {
            // f32 activations: vectorized load + pack to bf16 (full-K fast path)
            const int r = tid >> 1, cb = (tid & 1) * 16;
            const int grow = arowbase + r;
            float uf[16];
#pragma unroll
            for (int i = 0; i < 16; ++i) uf[i] = 0.0f;
            if (grow < M) {
                const float4* a4 =
                    (const float4*)((const float*)A + (size_t)grow * lda + k0 + cb);
                float4 u0 = a4[0], u1 = a4[1], u2 = a4[2], u3 = a4[3];
                uf[0]=u0.x; uf[1]=u0.y; uf[2]=u0.z; uf[3]=u0.w;
                uf[4]=u1.x; uf[5]=u1.y; uf[6]=u1.z; uf[7]=u1.w;
                uf[8]=u2.x; uf[9]=u2.y; uf[10]=u2.z; uf[11]=u2.w;
                uf[12]=u3.x; uf[13]=u3.y; uf[14]=u3.z; uf[15]=u3.w;
            }
            bf16x8 p0, p1;
#pragma unroll
            for (int i = 0; i < 8; ++i) { p0[i] = (__bf16)uf[i]; p1[i] = (__bf16)uf[8 + i]; }
            *(bf16x8*)&As[r][cb]       = p0;
            *(bf16x8*)(&As[r][cb] + 8) = p1;
        }
        if (issuer) __builtin_amdgcn_s_wait_tensorcnt(0);
        __syncthreads();
#else
        // Manual vectorized staging fallback
        {
            const int r = tid >> 1, cb = (tid & 1) * 16;
            const int grow = arowbase + r;
            if constexpr (ABF16) {
                bf16x8 v0 = {}, v1 = {};
                if (grow < M) {
                    const bf16x8* a8 =
                        (const bf16x8*)((const __bf16*)A + (size_t)grow * lda + k0 + cb);
                    v0 = a8[0]; v1 = a8[1];
                }
                *(bf16x8*)&As[r][cb]       = v0;
                *(bf16x8*)(&As[r][cb] + 8) = v1;
            } else {
                float uf[16];
#pragma unroll
                for (int i = 0; i < 16; ++i) uf[i] = 0.0f;
                if (grow < M) {
                    const float4* a4 =
                        (const float4*)((const float*)A + (size_t)grow * lda + k0 + cb);
                    float4 u0 = a4[0], u1 = a4[1], u2 = a4[2], u3 = a4[3];
                    uf[0]=u0.x; uf[1]=u0.y; uf[2]=u0.z; uf[3]=u0.w;
                    uf[4]=u1.x; uf[5]=u1.y; uf[6]=u1.z; uf[7]=u1.w;
                    uf[8]=u2.x; uf[9]=u2.y; uf[10]=u2.z; uf[11]=u2.w;
                    uf[12]=u3.x; uf[13]=u3.y; uf[14]=u3.z; uf[15]=u3.w;
                }
                bf16x8 p0, p1;
#pragma unroll
                for (int i = 0; i < 8; ++i) { p0[i] = (__bf16)uf[i]; p1[i] = (__bf16)uf[8+i]; }
                *(bf16x8*)&As[r][cb]       = p0;
                *(bf16x8*)(&As[r][cb] + 8) = p1;
            }
        }
        {
            const int n = tid >> 1, kb = (tid & 1) * 16;
            const int gcol = bcolbase + n;
            bf16x8 v0 = {}, v1 = {};
            if (gcol < N) {
                const bf16x8* b8 = (const bf16x8*)(Bt + (size_t)gcol * ldbt + k0 + kb);
                v0 = b8[0]; v1 = b8[1];
            }
            *(bf16x8*)&Bs[n][kb]       = v0;
            *(bf16x8*)(&Bs[n][kb] + 8) = v1;
        }
        __syncthreads();
#endif
        v16bf afr[2], bfr[4];
#pragma unroll
        for (int mi = 0; mi < 2; ++mi) afr[mi] = load_frag(&As[0][0], wm + mi * 16, lane);
#pragma unroll
        for (int ni = 0; ni < 4; ++ni) bfr[ni] = load_frag(&Bs[0][0], wn + ni * 16, lane);

#pragma unroll
        for (int mi = 0; mi < 2; ++mi)
#pragma unroll
            for (int ni = 0; ni < 4; ++ni)
                acc[mi][ni] = __builtin_amdgcn_wmma_f32_16x16x32_bf16(
                    false, afr[mi], false, bfr[ni], (short)0, acc[mi][ni],
                    false, false);
        __syncthreads();
    }
}

__device__ inline void store_tiles(
    float* __restrict__ C, int ldc, int M, int N,
    int rowbase, int colbase, const float* __restrict__ bias,
    v8f acc[2][4], int tid)
{
    const int lane = tid & 31;
    const int w    = tid >> 5;
    const int wm   = (w & 3) * 32;
    const int wn   = (w >> 2) * 64;
#pragma unroll
    for (int mi = 0; mi < 2; ++mi) {
#pragma unroll
        for (int ni = 0; ni < 4; ++ni) {
            const int r0 = rowbase + wm + mi * 16 + 8 * (lane >> 4);
            const int cc = colbase + wn + ni * 16 + (lane & 15);
            if (cc >= N) continue;
            const float badd = bias ? bias[cc] : 0.0f;
#pragma unroll
            for (int v = 0; v < 8; ++v) {
                const int r = r0 + v;
                if (r < M) C[(long)r * ldc + cc] = acc[mi][ni][v] + badd;
            }
        }
    }
}

// ---------------- generic GEMM: C = A[MxK] * Bt[NxK]^T (+bias) -------------
template<bool ABF16>
__global__ void __launch_bounds__(256)
gemm_wmma_kernel(const void* __restrict__ A, int lda,
                 const __bf16* __restrict__ Bt, int ldbt,
                 float* __restrict__ C, int ldc,
                 int M, int N, int K, const float* __restrict__ bias)
{
    __shared__ __bf16 As[128][LDSK];
    __shared__ __bf16 Bs[128][LDSK];
    const int tid = threadIdx.x;
    const int rowbase = blockIdx.y * 128;
    const int colbase = blockIdx.x * 128;
    v8f acc[2][4];
    const v8f zero = {0.f, 0.f, 0.f, 0.f, 0.f, 0.f, 0.f, 0.f};
#pragma unroll
    for (int mi = 0; mi < 2; ++mi)
#pragma unroll
        for (int ni = 0; ni < 4; ++ni) acc[mi][ni] = zero;

    gemm_tiles_accum<ABF16>(A, lda, M, rowbase, Bt, ldbt, N, colbase, K,
                            As, Bs, acc, tid);
    store_tiles(C, ldc, M, N, rowbase, colbase, bias, acc, tid);
}

// ------- gates = inp_use_n @ W_ih[n] + h_n @ W_hh[n] + b_lstm[n] -----------
__global__ void __launch_bounds__(256)
gates_wmma_kernel(const __bf16* __restrict__ inp_use,  // [NB][B][ATT] bf16
                  const __bf16* __restrict__ hpad,     // [B][NB*BSP] bf16 (zero-padded)
                  const __bf16* __restrict__ Wiht,     // [NB][G4][ATT]  (N x Kpad)
                  const __bf16* __restrict__ Whht,     // [NB][G4][BSP]
                  const float* __restrict__ b_lstm,    // [NB][G4]
                  float* __restrict__ gates)           // [NB][B][G4]
{
    __shared__ __bf16 As[128][LDSK];
    __shared__ __bf16 Bs[128][LDSK];
    const int tid = threadIdx.x;
    const int n = blockIdx.z;
    const int colbase = blockIdx.x * 128;
    v8f acc[2][4];
    const v8f zero = {0.f, 0.f, 0.f, 0.f, 0.f, 0.f, 0.f, 0.f};
#pragma unroll
    for (int mi = 0; mi < 2; ++mi)
#pragma unroll
        for (int ni = 0; ni < 4; ++ni) acc[mi][ni] = zero;

    gemm_tiles_accum<true>(inp_use + (size_t)n * B_ * ATT_, ATT_, B_, 0,
                           Wiht + (size_t)n * G4_ * ATT_, ATT_, G4_, colbase, ATT_,
                           As, Bs, acc, tid);
    gemm_tiles_accum<true>(hpad + n * BSP_, NB_ * BSP_, B_, 0,
                           Whht + (size_t)n * G4_ * BSP_, BSP_, G4_, colbase, BSP_,
                           As, Bs, acc, tid);

    store_tiles(gates + (size_t)n * B_ * G4_, G4_, B_, G4_,
                0, colbase, b_lstm + n * G4_, acc, tid);
}

// ---------------- small fused VALU kernels ----------------
__device__ inline float sigmoidf_(float x) { return 1.0f / (1.0f + __expf(-x)); }

__global__ void __launch_bounds__(256)
input_attn_kernel(const float* __restrict__ xt,   // [B][NINP]
                  const float* __restrict__ h,    // [B][NHID] f32 state
                  const float* __restrict__ Wq,   // [BS][DK]
                  const float* __restrict__ Wk,   // [NINP][DK]
                  const float* __restrict__ Wv,   // [NINP][ATT]
                  __bf16* __restrict__ inp_use,   // [NB][B][ATT] bf16
                  float* __restrict__ mask_ws,    // [B][NB]
                  float* __restrict__ mask_out)   // d_out masks slot [B][NB]
{
    const int b = blockIdx.x, t = threadIdx.x;
    __shared__ float xs[NINP_];
    __shared__ float hs[NHID_];
    __shared__ float kx[DK_];
    __shared__ float qv[NB_ * DK_];
    __shared__ float attn[NB_];
    __shared__ float msk[NB_];

    for (int i = t; i < NINP_; i += 256) xs[i] = xt[(long)b * NINP_ + i];
    for (int i = t; i < NHID_; i += 256) hs[i] = h[(long)b * NHID_ + i];
    __syncthreads();

    if (t < DK_) {
        float s = 0.f;
        for (int d = 0; d < NINP_; ++d) s += xs[d] * Wk[d * DK_ + t];
        kx[t] = s;
    }
    for (int o = t; o < NB_ * DK_; o += 256) {
        const int n = o >> 6, k = o & 63;
        float s = 0.f;
        for (int d = 0; d < BS_; ++d) s += hs[n * BS_ + d] * Wq[d * DK_ + k];
        qv[o] = s;
    }
    __syncthreads();
    if (t < NB_) {
        float s = 0.f;
        for (int k = 0; k < DK_; ++k) s += qv[t * DK_ + k] * kx[k];
        attn[t] = sigmoidf_(s * 0.125f);   // 1/sqrt(DK)
    }
    __syncthreads();
    if (t == 0) {  // top-4 of 6: drop two smallest (late index on ties)
        float a[NB_], mk[NB_];
        for (int i = 0; i < NB_; ++i) { a[i] = attn[i]; mk[i] = 1.0f; }
        for (int drop = 0; drop < NB_ - 4; ++drop) {
            int mi = 0; float mv = 1e30f;
            for (int i = 0; i < NB_; ++i)
                if (mk[i] > 0.5f && a[i] <= mv) { mv = a[i]; mi = i; }
            mk[mi] = 0.0f;
        }
        for (int i = 0; i < NB_; ++i) msk[i] = mk[i];
    }
    __syncthreads();
    if (t < NB_) {
        mask_ws[b * NB_ + t]  = msk[t];
        mask_out[b * NB_ + t] = msk[t];
    }
    for (int v = t; v < ATT_; v += 256) {
        float s = 0.f;
        for (int d = 0; d < NINP_; ++d) s += xs[d] * Wv[d * ATT_ + v];
#pragma unroll
        for (int n = 0; n < NB_; ++n)
            inp_use[((size_t)n * B_ + b) * ATT_ + v] = (__bf16)(attn[n] * s);
    }
}

__global__ void __launch_bounds__(256)
lstm_kernel(const float* __restrict__ gates,   // [NB][B][G4]
            const float* __restrict__ h_in,    // [B][NHID]
            const float* __restrict__ mask,    // [B][NB]
            float* __restrict__ c_state,       // [B][NHID] in/out
            float* __restrict__ h_upd,         // [B][NHID] f32
            __bf16* __restrict__ h_upd_pad,    // [B*NB][BSP] bf16 (pads stay 0)
            float* __restrict__ cT_out)        // d_out c_T slot
{
    const int idx = blockIdx.x * 256 + threadIdx.x;
    if (idx >= B_ * NHID_) return;
    const int b = idx / NHID_;
    const int r = idx - b * NHID_;
    const int n = r / BS_;
    const int j = r - n * BS_;
    const float* g = gates + ((size_t)n * B_ + b) * G4_;
    const float ig = sigmoidf_(g[j]);
    const float fg = sigmoidf_(g[BS_ + j]);
    const float gg = tanhf(g[2 * BS_ + j]);
    const float og = sigmoidf_(g[3 * BS_ + j]);
    const float c_old = c_state[idx];
    const float c_new = fg * c_old + ig * gg;
    const float h_new = og * tanhf(c_new);
    const float m = mask[b * NB_ + n];
    const float cu = m * c_new + (1.0f - m) * c_old;
    const float hu = m * h_new + (1.0f - m) * h_in[idx];
    c_state[idx] = cu;
    cT_out[idx]  = cu;
    h_upd[idx]   = hu;
    h_upd_pad[((size_t)b * NB_ + n) * BSP_ + j] = (__bf16)hu;
}

__global__ void __launch_bounds__(256)
comm_attn_kernel(const float* __restrict__ qkv,  // [B*NB][384]
                 __bf16* __restrict__ ctx)       // [B*NB][128] bf16
{
    const int b = blockIdx.x, t = threadIdx.x;
    __shared__ float sc[NB_ * NB_];
    __shared__ float att[NB_ * NB_];
    const float inv = 0.1767766953f; // 1/sqrt(DKC)
    for (int hh = 0; hh < HC_; ++hh) {
        if (t < NB_ * NB_) {
            const int n = t / NB_, m = t % NB_;
            const float* qp = qkv + ((size_t)(b * NB_ + n)) * 384 + hh * DKC_;
            const float* kp = qkv + ((size_t)(b * NB_ + m)) * 384 + 128 + hh * DKC_;
            float s = 0.f;
#pragma unroll
            for (int k = 0; k < DKC_; ++k) s += qp[k] * kp[k];
            sc[t] = s * inv;
        }
        __syncthreads();
        if (t < NB_) {
            float mx = -1e30f;
            for (int m = 0; m < NB_; ++m) mx = fmaxf(mx, sc[t * NB_ + m]);
            float e[NB_], den = 0.f;
            for (int m = 0; m < NB_; ++m) { e[m] = __expf(sc[t * NB_ + m] - mx); den += e[m]; }
            const float rd = 1.0f / den;
            for (int m = 0; m < NB_; ++m) att[t * NB_ + m] = e[m] * rd;
        }
        __syncthreads();
        if (t < NB_ * DVC_) {
            const int n = t >> 5, d = t & 31;
            float s = 0.f;
#pragma unroll
            for (int m = 0; m < NB_; ++m)
                s += att[n * NB_ + m] * qkv[((size_t)(b * NB_ + m)) * 384 + 256 + hh * DVC_ + d];
            ctx[((size_t)(b * NB_ + n)) * 128 + hh * DVC_ + d] = (__bf16)s;
        }
        __syncthreads();
    }
}

__global__ void __launch_bounds__(256)
combine_kernel(const float* __restrict__ h_upd,
               const float* __restrict__ outtmp,  // [B][NHID] view of [768][200]
               const float* __restrict__ mask,
               float* __restrict__ h_state,       // f32
               __bf16* __restrict__ hpad,         // [B][NB*BSP] bf16
               float* __restrict__ outs_slot,
               float* __restrict__ hT_slot)
{
    const int idx = blockIdx.x * 256 + threadIdx.x;
    if (idx >= B_ * NHID_) return;
    const int b = idx / NHID_;
    const int r = idx - b * NHID_;
    const int n = r / BS_;
    const int j = r - n * BS_;
    const float v = h_upd[idx] + mask[b * NB_ + n] * outtmp[idx];
    h_state[idx]   = v;
    outs_slot[idx] = v;
    hT_slot[idx]   = v;
    hpad[(size_t)b * (NB_ * BSP_) + n * BSP_ + j] = (__bf16)v;
}

// state init: h/c f32 state + zero-padded bf16 h, pads zeroed
__global__ void init_state_kernel(const float* __restrict__ hx0,
                                  const float* __restrict__ cx0,
                                  float* __restrict__ h, float* __restrict__ c,
                                  __bf16* __restrict__ hpad,
                                  __bf16* __restrict__ h_upd_pad)
{
    const int i = blockIdx.x * 256 + threadIdx.x;
    if (i >= B_ * NB_ * BSP_) return;
    const int b = i / (NB_ * BSP_);
    const int r = i - b * (NB_ * BSP_);
    const int n = r / BSP_;
    const int j = r - n * BSP_;
    if (j < BS_) {
        const int idx = b * NHID_ + n * BS_ + j;
        const float hv = hx0[idx];
        h[idx] = hv;
        c[idx] = cx0[idx];
        hpad[i] = (__bf16)hv;
    } else {
        hpad[i] = (__bf16)0.0f;
    }
    h_upd_pad[i] = (__bf16)0.0f;   // lstm only writes j<BS each step; pads stay 0
}

// transpose + f32->bf16 + zero-pad K: W[K][N] -> Wt[N][Kpad], batched over z
__global__ void transpose_conv_kernel(const float* __restrict__ W,
                                      int K, int N, int Kpad,
                                      long srcBatchStride, long dstBatchStride,
                                      __bf16* __restrict__ Wt)
{
    const long z = blockIdx.z;
    const long idx = (long)blockIdx.x * 256 + threadIdx.x;
    if (idx >= (long)N * Kpad) return;
    const int n = (int)(idx / Kpad);
    const int k = (int)(idx - (long)n * Kpad);
    const float v = (k < K) ? W[z * srcBatchStride + (long)k * N + n] : 0.0f;
    Wt[z * dstBatchStride + idx] = (__bf16)v;
}

__global__ void concat_qkvw_kernel(const float* __restrict__ Wq,
                                   const float* __restrict__ Wk,
                                   const float* __restrict__ Wv,
                                   float* __restrict__ W) // [BS][384]
{
    const int i = blockIdx.x * 256 + threadIdx.x;
    if (i >= BS_ * 384) return;
    const int r = i / 384, c = i - r * 384;
    float v;
    if (c < 128)      v = Wq[r * 128 + c];
    else if (c < 256) v = Wk[r * 128 + (c - 128)];
    else              v = Wv[r * 128 + (c - 256)];
    W[i] = v;
}

// ---------------------------------------------------------------------------
extern "C" void kernel_launch(void* const* d_in, const int* in_sizes, int n_in,
                              void* d_out, int out_size, void* d_ws, size_t ws_size,
                              hipStream_t stream) {
    const float* input = (const float*)d_in[0];
    const float* hx0   = (const float*)d_in[1];
    const float* cx0   = (const float*)d_in[2];
    const float* enc_W = (const float*)d_in[3];
    const float* enc_b = (const float*)d_in[4];
    const float* Wq_i  = (const float*)d_in[5];
    const float* Wk_i  = (const float*)d_in[6];
    const float* Wv_i  = (const float*)d_in[7];
    const float* W_ih  = (const float*)d_in[8];
    const float* W_hh  = (const float*)d_in[9];
    const float* b_ls  = (const float*)d_in[10];
    const float* Wq_c  = (const float*)d_in[11];
    const float* Wk_c  = (const float*)d_in[12];
    const float* Wv_c  = (const float*)d_in[13];
    const float* Wo_c  = (const float*)d_in[14];
    float* out = (float*)d_out;

    const long OFF_HT    = (long)T_ * B_ * NHID_;
    const long OFF_CT    = OFF_HT + (long)B_ * NHID_;
    const long OFF_MASKS = OFF_CT + (long)B_ * NHID_;

    char* base = (char*)d_ws;
    size_t off = 0;
    auto carve = [&](size_t bytes) -> char* {
        char* p = base + off;
        off = (off + bytes + 255) & ~(size_t)255;
        return p;
    };
    float*  ws_emb      = (float*) carve((size_t)T_ * B_ * NINP_ * 4);   // 78.6 MB
    __bf16* ws_encWt    = (__bf16*)carve((size_t)NINP_ * NTOK_ * 2);     // [600][1024]
    __bf16* ws_Wiht     = (__bf16*)carve((size_t)NB_ * G4_ * ATT_ * 2);  // [6][800][512]
    __bf16* ws_Whht     = (__bf16*)carve((size_t)NB_ * G4_ * BSP_ * 2);  // [6][800][224]
    float*  ws_Wqkv     = (float*) carve((size_t)BS_ * 384 * 4);
    __bf16* ws_Wqkvt    = (__bf16*)carve((size_t)384 * BSP_ * 2);        // [384][224]
    __bf16* ws_Wot      = (__bf16*)carve((size_t)BS_ * 128 * 2);         // [200][128]
    __bf16* ws_inp_use  = (__bf16*)carve((size_t)NB_ * B_ * ATT_ * 2);
    float*  ws_mask     = (float*) carve((size_t)B_ * NB_ * 4);
    float*  ws_gates    = (float*) carve((size_t)NB_ * B_ * G4_ * 4);
    float*  ws_h        = (float*) carve((size_t)B_ * NHID_ * 4);
    __bf16* ws_hpad     = (__bf16*)carve((size_t)B_ * NB_ * BSP_ * 2);
    float*  ws_c        = (float*) carve((size_t)B_ * NHID_ * 4);
    float*  ws_hupd     = (float*) carve((size_t)B_ * NHID_ * 4);
    __bf16* ws_hupd_pad = (__bf16*)carve((size_t)B_ * NB_ * BSP_ * 2);
    float*  ws_qkv      = (float*) carve((size_t)B_ * NB_ * 384 * 4);
    __bf16* ws_ctx      = (__bf16*)carve((size_t)B_ * NB_ * 128 * 2);
    float*  ws_outtmp   = (float*) carve((size_t)B_ * NB_ * BS_ * 4);
    (void)ws_size; (void)in_sizes; (void)n_in; (void)out_size;

    // ---- one-time per call: state init + weight transpose/convert ----
    init_state_kernel<<<(B_ * NB_ * BSP_ + 255) / 256, 256, 0, stream>>>(
        hx0, cx0, ws_h, ws_c, ws_hpad, ws_hupd_pad);
    concat_qkvw_kernel<<<(BS_ * 384 + 255) / 256, 256, 0, stream>>>(
        Wq_c, Wk_c, Wv_c, ws_Wqkv);

    {   // enc_W [1024][600] -> [600][1024]
        dim3 g(((long)NINP_ * NTOK_ + 255) / 256, 1, 1);
        transpose_conv_kernel<<<g, 256, 0, stream>>>(enc_W, NTOK_, NINP_, NTOK_,
                                                     0, 0, ws_encWt);
    }
    {   // W_ih [6][512][800] -> [6][800][512]
        dim3 g(((long)G4_ * ATT_ + 255) / 256, 1, NB_);
        transpose_conv_kernel<<<g, 256, 0, stream>>>(W_ih, ATT_, G4_, ATT_,
                                                     (long)ATT_ * G4_, (long)G4_ * ATT_, ws_Wiht);
    }
    {   // W_hh [6][200][800] -> [6][800][224]
        dim3 g(((long)G4_ * BSP_ + 255) / 256, 1, NB_);
        transpose_conv_kernel<<<g, 256, 0, stream>>>(W_hh, BS_, G4_, BSP_,
                                                     (long)BS_ * G4_, (long)G4_ * BSP_, ws_Whht);
    }
    {   // Wqkv [200][384] -> [384][224]
        dim3 g(((long)384 * BSP_ + 255) / 256, 1, 1);
        transpose_conv_kernel<<<g, 256, 0, stream>>>(ws_Wqkv, BS_, 384, BSP_,
                                                     0, 0, ws_Wqkvt);
    }
    {   // Wo_c [128][200] -> [200][128]
        dim3 g(((long)BS_ * 128 + 255) / 256, 1, 1);
        transpose_conv_kernel<<<g, 256, 0, stream>>>(Wo_c, 128, BS_, 128,
                                                     0, 0, ws_Wot);
    }

    // ---- encoder: emb = input @ enc_W + enc_b ----
    {
        dim3 grid((NINP_ + 127) / 128, (T_ * B_ + 127) / 128);
        gemm_wmma_kernel<false><<<grid, 256, 0, stream>>>(
            input, NTOK_, ws_encWt, NTOK_, ws_emb, NINP_,
            T_ * B_, NINP_, NTOK_, enc_b);
    }

    // ---- recurrent scan ----
    for (int t = 0; t < T_; ++t) {
        const float* xt = ws_emb + (long)t * B_ * NINP_;

        input_attn_kernel<<<B_, 256, 0, stream>>>(
            xt, ws_h, Wq_i, Wk_i, Wv_i, ws_inp_use, ws_mask,
            out + OFF_MASKS + (long)t * B_ * NB_);

        {
            dim3 grid((G4_ + 127) / 128, 1, NB_);
            gates_wmma_kernel<<<grid, 256, 0, stream>>>(
                ws_inp_use, ws_hpad, ws_Wiht, ws_Whht, b_ls, ws_gates);
        }

        lstm_kernel<<<(B_ * NHID_ + 255) / 256, 256, 0, stream>>>(
            ws_gates, ws_h, ws_mask, ws_c, ws_hupd, ws_hupd_pad, out + OFF_CT);

        {   // fused q/k/v projection: [768x224]@[224x384]
            dim3 grid((384 + 127) / 128, (B_ * NB_ + 127) / 128);
            gemm_wmma_kernel<true><<<grid, 256, 0, stream>>>(
                ws_hupd_pad, BSP_, ws_Wqkvt, BSP_, ws_qkv, 384,
                B_ * NB_, 384, BSP_, nullptr);
        }

        comm_attn_kernel<<<B_, 256, 0, stream>>>(ws_qkv, ws_ctx);

        {   // output projection: [768x128]@[128x200]
            dim3 grid((BS_ + 127) / 128, (B_ * NB_ + 127) / 128);
            gemm_wmma_kernel<true><<<grid, 256, 0, stream>>>(
                ws_ctx, 128, ws_Wot, 128, ws_outtmp, BS_,
                B_ * NB_, BS_, 128, nullptr);
        }

        combine_kernel<<<(B_ * NHID_ + 255) / 256, 256, 0, stream>>>(
            ws_hupd, ws_outtmp, ws_mask, ws_h, ws_hpad,
            out + (long)t * B_ * NHID_, out + OFF_HT);
    }
}
